// BipartateMatching_17549236371825
// MI455X (gfx1250) — compile-verified, matrix operand order checked by reading, not run on hardware
//
#include <hip/hip_runtime.h>
#include <hip/hip_bf16.h>
#include <math.h>

#define N 4096
#define DFEAT 64
#define ROUNDS 128

typedef float v2f  __attribute__((ext_vector_type(2)));
typedef float v8f  __attribute__((ext_vector_type(8)));
typedef _Float16 v16h __attribute__((ext_vector_type(16)));

#ifdef __has_builtin
#  if __has_builtin(__builtin_amdgcn_wmma_f32_16x16x4_f32)
#    define HAVE_WMMA_F32X4 1
#  endif
#endif
#ifndef HAVE_WMMA_F32X4
#  define HAVE_WMMA_F32X4 0
#endif

// ---------------------------------------------------------------------------
// Init: squared norms + clear match state.
// ---------------------------------------------------------------------------
__global__ __launch_bounds__(256) void bm_init_kernel(
    const float* __restrict__ X, const float* __restrict__ Y,
    float* __restrict__ x2, float* __restrict__ y2,
    int* __restrict__ rowMatch, int* __restrict__ colMatch) {
  int i = blockIdx.x * blockDim.x + threadIdx.x;
  if (i >= N) return;
  float sx = 0.f, sy = 0.f;
#pragma unroll 8
  for (int k = 0; k < DFEAT; ++k) {
    float a = X[i * DFEAT + k];
    float b = Y[i * DFEAT + k];
    sx += a * a;
    sy += b * b;
  }
  x2[i] = sx;
  y2[i] = sy;
  rowMatch[i] = -1;
  colMatch[i] = -1;
}

// ---------------------------------------------------------------------------
// Cost matrix via WMMA: one wave32 per 16x16 tile of cost.
// cost[i][j] = sqrt(max(x2[i] + y2[j] - 2 * dot(X_i, Y_j), 0))
// ---------------------------------------------------------------------------
__global__ __launch_bounds__(32) void bm_cost_tile_kernel(
    const float* __restrict__ X, const float* __restrict__ Y,
    const float* __restrict__ x2, const float* __restrict__ y2,
    float* __restrict__ cost) {
  const int m0   = blockIdx.y * 16;
  const int n0   = blockIdx.x * 16;
  const int lane = threadIdx.x;   // 0..31, full wave (EXEC all ones)
  const int l16  = lane & 15;
  const int half = lane >> 4;

  v8f acc = {};

#if HAVE_WMMA_F32X4
  // f32 WMMA 16x16x4, 16 k-steps over DFEAT=64.
  // A 16x4 layout: lanes 0-15 hold K0,K1 in v0,v1; lanes 16-31 hold K2,K3.
  // B 4x16 assumed mirrored (row K striped across lanes).
  const float* arow = X + (size_t)(m0 + l16) * DFEAT + half * 2;
  const float* brow = Y + (size_t)(n0 + l16) * DFEAT + half * 2;
#pragma unroll
  for (int kk = 0; kk < DFEAT / 4; ++kk) {
    v2f a = *(const v2f*)(arow + kk * 4);
    v2f b = *(const v2f*)(brow + kk * 4);
    acc = __builtin_amdgcn_wmma_f32_16x16x4_f32(
        /*neg_a=*/false, a, /*neg_b=*/false, b,
        /*c_mod=*/(short)0, acc, /*reuse_a=*/false, /*reuse_b=*/false);
  }
#else
  // Fallback: f16 WMMA 16x16x32, two k-steps over DFEAT=64.
#pragma unroll
  for (int kk = 0; kk < 2; ++kk) {
    v16h a, b;
#pragma unroll
    for (int e = 0; e < 16; ++e) {
      int r  = e >> 1;  // A VGPR index 0..7
      int kb = (r < 4 ? r * 2 : 16 + (r - 4) * 2) + (half ? 8 : 0) + (e & 1) + kk * 32;
      a[e] = (_Float16)X[(size_t)(m0 + l16) * DFEAT + kb];
      b[e] = (_Float16)Y[(size_t)(n0 + l16) * DFEAT + kb];
    }
    acc = __builtin_amdgcn_wmma_f32_16x16x32_f16(
        false, a, false, b, (short)0, acc, false, false);
  }
#endif

  // D layout: VGPR v, lanes 0-15 -> M=v, N=lane; lanes 16-31 -> M=v+8, N=lane-16
  const int   j  = n0 + l16;
  const float yj = y2[j];
#pragma unroll
  for (int v = 0; v < 8; ++v) {
    int   i = m0 + v + half * 8;
    float d = x2[i] + yj - 2.0f * acc[v];
    cost[(size_t)i * N + j] = sqrtf(fmaxf(d, 0.0f));
  }
}

// ---------------------------------------------------------------------------
// Per-round: row argmin over unmatched columns (one block per row).
// Tie-break: smaller j (matches stable flat-index argsort order).
// ---------------------------------------------------------------------------
__global__ __launch_bounds__(256) void bm_row_argmin_kernel(
    const float* __restrict__ cost, const int* __restrict__ rowMatch,
    const int* __restrict__ colMatch, int* __restrict__ rowIdx) {
  const int i = blockIdx.x;
  __shared__ float sval[256];
  __shared__ int   sidx[256];
  if (rowMatch[i] >= 0) {
    if (threadIdx.x == 0) rowIdx[i] = -1;
    return;
  }
  float best = INFINITY;
  int   bj   = 0x7fffffff;
  const float* row = cost + (size_t)i * N;
  for (int j = threadIdx.x; j < N; j += 256) {
    if (colMatch[j] < 0) {
      float c = row[j];
      if (c < best || (c == best && j < bj)) { best = c; bj = j; }
    }
  }
  sval[threadIdx.x] = best;
  sidx[threadIdx.x] = bj;
  __syncthreads();
  for (int s = 128; s > 0; s >>= 1) {
    if (threadIdx.x < s) {
      float ov = sval[threadIdx.x + s];
      int   oj = sidx[threadIdx.x + s];
      if (ov < sval[threadIdx.x] ||
          (ov == sval[threadIdx.x] && oj < sidx[threadIdx.x])) {
        sval[threadIdx.x] = ov;
        sidx[threadIdx.x] = oj;
      }
    }
    __syncthreads();
  }
  if (threadIdx.x == 0)
    rowIdx[i] = (sidx[0] == 0x7fffffff) ? -1 : sidx[0];
}

// ---------------------------------------------------------------------------
// Per-round: column argmin over unmatched rows (one thread per column;
// coalesced across the row-major cost matrix). Strict < keeps smallest i
// on ties (stable order).
// ---------------------------------------------------------------------------
__global__ __launch_bounds__(256) void bm_col_argmin_kernel(
    const float* __restrict__ cost, const int* __restrict__ rowMatch,
    const int* __restrict__ colMatch, int* __restrict__ colIdx) {
  const int j = blockIdx.x * 256 + threadIdx.x;
  if (colMatch[j] >= 0) { colIdx[j] = -1; return; }
  float best = INFINITY;
  int   bi   = -1;
  for (int i = 0; i < N; ++i) {
    if (rowMatch[i] < 0) {
      float c = cost[(size_t)i * N + j];
      if (c < best) { best = c; bi = i; }
    }
  }
  colIdx[j] = bi;
}

// ---------------------------------------------------------------------------
// Per-round: commit mutual minima.
// ---------------------------------------------------------------------------
__global__ __launch_bounds__(256) void bm_match_kernel(
    const int* __restrict__ rowIdx, const int* __restrict__ colIdx,
    int* __restrict__ rowMatch, int* __restrict__ colMatch) {
  int i = blockIdx.x * blockDim.x + threadIdx.x;
  if (i >= N) return;
  if (rowMatch[i] >= 0) return;
  int j = rowIdx[i];
  if (j >= 0 && colIdx[j] == i) {
    rowMatch[i] = j;  // unique writer: colIdx[j]==i for exactly one i
    colMatch[j] = i;
  }
}

__global__ __launch_bounds__(256) void bm_finalize_kernel(
    const int* __restrict__ rowMatch, int* __restrict__ out) {
  int i = blockIdx.x * blockDim.x + threadIdx.x;
  if (i < N) out[i] = rowMatch[i];
}

// ---------------------------------------------------------------------------
extern "C" void kernel_launch(void* const* d_in, const int* in_sizes, int n_in,
                              void* d_out, int out_size, void* d_ws, size_t ws_size,
                              hipStream_t stream) {
  const float* X = (const float*)d_in[0];
  const float* Y = (const float*)d_in[1];
  int* out = (int*)d_out;  // reference output dtype is int32

  char*  ws  = (char*)d_ws;
  size_t off = 0;
  float* cost = (float*)(ws + off); off += (size_t)N * N * sizeof(float);
  float* x2   = (float*)(ws + off); off += (size_t)N * sizeof(float);
  float* y2   = (float*)(ws + off); off += (size_t)N * sizeof(float);
  int* rowIdx   = (int*)(ws + off); off += (size_t)N * sizeof(int);
  int* colIdx   = (int*)(ws + off); off += (size_t)N * sizeof(int);
  int* rowMatch = (int*)(ws + off); off += (size_t)N * sizeof(int);
  int* colMatch = (int*)(ws + off); off += (size_t)N * sizeof(int);

  bm_init_kernel<<<N / 256, 256, 0, stream>>>(X, Y, x2, y2, rowMatch, colMatch);

  dim3 cgrid(N / 16, N / 16);
  bm_cost_tile_kernel<<<cgrid, 32, 0, stream>>>(X, Y, x2, y2, cost);

  for (int r = 0; r < ROUNDS; ++r) {
    bm_row_argmin_kernel<<<N, 256, 0, stream>>>(cost, rowMatch, colMatch, rowIdx);
    bm_col_argmin_kernel<<<N / 256, 256, 0, stream>>>(cost, rowMatch, colMatch, colIdx);
    bm_match_kernel<<<N / 256, 256, 0, stream>>>(rowIdx, colIdx, rowMatch, colMatch);
  }

  bm_finalize_kernel<<<N / 256, 256, 0, stream>>>(rowMatch, out);
}
